// globalfeaturefuse_6485400617413
// MI455X (gfx1250) — matrix operand order repaired and checked by reading, not compiled, not run on hardware
//
#include <hip/hip_runtime.h>
#include <hip/hip_bf16.h>
#include <math.h>

// ---------------------------------------------------------------------------
// globalfeaturefuse for MI455X (gfx1250):
//   q = x@Wq^T+bq ; k = y@Wk^T+bk ; v = y@Wv^T+bv        (bf16 WMMA GEMMs)
//   per-token 8x8 head-attention softmax                  (VALU, tiny)
//   out = attn@Wo^T + bo + x                              (bf16 WMMA GEMM, f32 out)
// GEMM tiles staged to LDS by the Tensor Data Mover (tensor_load_to_lds),
// double-buffered so the TDM overlaps V_WMMA_F32_16X16X32_BF16 compute.
// Wave tile 64x64 -> 16 WMMAs per 16 ds_load_b128 per K-step.
// ---------------------------------------------------------------------------

typedef __bf16 bf16;
typedef __attribute__((ext_vector_type(8)))  bf16  v8bf;
typedef __attribute__((ext_vector_type(16))) bf16  v16bf;
typedef __attribute__((ext_vector_type(8)))  float v8f;
typedef __attribute__((ext_vector_type(4)))  unsigned int u32x4;
typedef __attribute__((ext_vector_type(4)))  int i32x4;
typedef __attribute__((ext_vector_type(8)))  int i32x8;

#define T_TOK  8192
#define C_DIM  2048
#define H_HEADS 8
#define D_HEAD  256
#define ATT_SCALE 0.0625f   // 256^-0.5

// Block tile 128(M) x 256(N) x 32(K); 8 waves in 2x4; wave tile 64x64.
#define BM 128
#define BN 256
#define BK 32

// ---------------------------------------------------------------------------
// TDM 2D tile load: copy tile_rows x (tile_row_dwords*4B) from a row-major
// matrix (row pitch = row_dwords dwords) into LDS, rows packed contiguously.
// D# built per CDNA5 ISA section 8 (groups 0..3), data_size = 4 bytes.
// ---------------------------------------------------------------------------
__device__ __forceinline__ void tdm_load_tile_2d(unsigned lds_byte_off,
                                                 const bf16* gsrc,
                                                 unsigned tensor_rows,
                                                 unsigned row_dwords,
                                                 unsigned tile_rows,
                                                 unsigned tile_row_dwords)
{
    unsigned long long ga = (unsigned long long)(uintptr_t)gsrc;
    u32x4 g0;
    g0[0] = 1u;                                           // count=1 (valid), no gather
    g0[1] = lds_byte_off;                                 // lds_addr
    g0[2] = (unsigned)(ga & 0xffffffffu);                 // global_addr[31:0]
    g0[3] = (unsigned)((ga >> 32) & 0x01ffffffu)          // global_addr[56:32]
          | 0x80000000u;                                  // type=2 ("image")
    i32x8 g1;
    g1[0] = (int)(2u << 16);                              // data_size=4B; mask/pad/iter=0
    g1[1] = (int)((row_dwords & 0xffffu) << 16);          // tensor_dim0[15:0]
    g1[2] = (int)((row_dwords >> 16)
          | ((tensor_rows & 0xffffu) << 16));             // tensor_dim0 hi | tensor_dim1 lo
    g1[3] = (int)((tensor_rows >> 16)
          | ((tile_row_dwords & 0xffffu) << 16));         // tensor_dim1 hi | tile_dim0
    g1[4] = (int)(tile_rows & 0xffffu);                   // tile_dim1 (tile_dim2=0)
    g1[5] = (int)row_dwords;                              // tensor_dim0_stride[31:0]
    g1[6] = 0;                                            // stride hi | dim1_stride lo
    g1[7] = 0;
    i32x4 g2 = {0, 0, 0, 0};                              // 2D tile: dims 2..4 unused
    i32x4 g3 = {0, 0, 0, 0};
#if defined(__clang_major__) && __clang_major__ >= 23
    i32x8 g4 = {0, 0, 0, 0, 0, 0, 0, 0};
    __builtin_amdgcn_tensor_load_to_lds(g0, g1, g2, g3, g4, 0);
#else
    __builtin_amdgcn_tensor_load_to_lds(g0, g1, g2, g3, 0);
#endif
}

// ---------------------------------------------------------------------------
// f32 -> bf16 elementwise conversion
// ---------------------------------------------------------------------------
__global__ __launch_bounds__(256)
void cvt_f32_to_bf16(const float* __restrict__ in, bf16* __restrict__ out, int n)
{
    int i = blockIdx.x * blockDim.x + threadIdx.x;
    int stride = gridDim.x * blockDim.x;
    for (; i < n; i += stride)
        out[i] = (bf16)in[i];
}

// ---------------------------------------------------------------------------
// GEMM: out[m,n] = sum_k A[m,k]*B[n,k] + bias[n] (+ residual[m,n])
// A: [M,K] bf16 row-major, B: [N,K] bf16 row-major.
// M multiple of 128, N multiple of 256, K multiple of 32.
// 256 threads = 8 waves (2x4); wave tile 64x64 = 4x4 fragments of 16x16.
// Double-buffered TDM staging: next tile's DMA in flight during WMMA.
// ---------------------------------------------------------------------------
__global__ __launch_bounds__(256)
void gemm_bf16_wmma(const bf16* __restrict__ A, const bf16* __restrict__ B,
                    const float* __restrict__ bias,
                    bf16* __restrict__ outB, float* __restrict__ outF,
                    const float* __restrict__ residual,
                    int M, int N, int K)
{
    __shared__ bf16 sA[2][BM * BK];   // 2 x 8 KB,  [row][k] 64B rows
    __shared__ bf16 sB[2][BN * BK];   // 2 x 16 KB, [n][k]   64B rows

    const int tid  = threadIdx.x;
    const int lane = tid & 31;
    const int wave = tid >> 5;
    const int wm   = wave >> 2;          // 0..1 -> 64 rows each
    const int wn   = wave & 3;           // 0..3 -> 64 cols each
    const int bm   = blockIdx.y * BM;
    const int bn   = blockIdx.x * BN;
    const int lrow = lane & 15;
    const int half = lane >> 4;

    const unsigned ldsA0 = (unsigned)(uintptr_t)(void*)&sA[0][0];
    const unsigned ldsA1 = (unsigned)(uintptr_t)(void*)&sA[1][0];
    const unsigned ldsB0 = (unsigned)(uintptr_t)(void*)&sB[0][0];
    const unsigned ldsB1 = (unsigned)(uintptr_t)(void*)&sB[1][0];

    v8f acc[4][4];
#pragma unroll
    for (int fm = 0; fm < 4; ++fm)
#pragma unroll
        for (int fn = 0; fn < 4; ++fn) {
            v8f z = {0.f, 0.f, 0.f, 0.f, 0.f, 0.f, 0.f, 0.f};
            acc[fm][fn] = z;
        }

    const unsigned rowDw = (unsigned)(K >> 1);       // bf16 row -> dwords
    const bf16* Abase = A + (size_t)bm * K;
    const bf16* Bbase = B + (size_t)bn * K;
    const int   nk    = K / BK;

    // Prologue: stage tile 0 into buffer 0.
    if (wave == 0) {
        tdm_load_tile_2d(ldsA0, Abase, (unsigned)M, rowDw, (unsigned)BM, 16u);
        tdm_load_tile_2d(ldsB0, Bbase, (unsigned)N, rowDw, (unsigned)BN, 16u);
    }

    for (int it = 0; it < nk; ++it) {
        const int buf = it & 1;
        __syncthreads();   // all waves done reading buf^1 (tile it-1)
        if (wave == 0) {
            if (it + 1 < nk) {
                const int koff = (it + 1) * BK;      // bf16 elements
                tdm_load_tile_2d(buf ? ldsA0 : ldsA1, Abase + koff,
                                 (unsigned)M, rowDw, (unsigned)BM, 16u);
                tdm_load_tile_2d(buf ? ldsB0 : ldsB1, Bbase + koff,
                                 (unsigned)N, rowDw, (unsigned)BN, 16u);
                __builtin_amdgcn_s_wait_tensorcnt(2);   // tile `it` done, next in flight
            } else {
                __builtin_amdgcn_s_wait_tensorcnt(0);   // final tile done
            }
        }
        __syncthreads();   // tile `it` visible to all waves

        // B fragments (32x16 bf16): lane = column n; lanes 0-15 K=0..15,
        // lanes 16-31 K=16..31, two values per VGPR (contiguous k order, as for A).
        v16bf bfr[4];
#pragma unroll
        for (int fn = 0; fn < 4; ++fn) {
            const bf16* p = &sB[buf][(wn * 64 + fn * 16 + lrow) * BK + half * 16];
            v8bf lo = *(const v8bf*)(p);
            v8bf hi = *(const v8bf*)(p + 8);
            bfr[fn] = __builtin_shufflevector(lo, hi,
                0, 1, 2, 3, 4, 5, 6, 7, 8, 9, 10, 11, 12, 13, 14, 15);
        }
        // A fragments (16x32 bf16): lanes 0-15 hold K 0..15,
        // lanes 16-31 hold K 16..31.
#pragma unroll
        for (int fm = 0; fm < 4; ++fm) {
            const bf16* p = &sA[buf][(wm * 64 + fm * 16 + lrow) * BK];
            v8bf lo = *(const v8bf*)(p + half * 16);
            v8bf hi = *(const v8bf*)(p + half * 16 + 8);
            v16bf afr = __builtin_shufflevector(lo, hi,
                0, 1, 2, 3, 4, 5, 6, 7, 8, 9, 10, 11, 12, 13, 14, 15);
#pragma unroll
            for (int fn = 0; fn < 4; ++fn)
                acc[fm][fn] = __builtin_amdgcn_wmma_f32_16x16x32_bf16(
                    false, afr, false, bfr[fn],
                    (short)0, acc[fm][fn], false, false);
        }
    }

    // Epilogue: C/D layout — VGPR r: lanes 0-15 -> M=r, lanes 16-31 -> M=8+r.
#pragma unroll
    for (int fm = 0; fm < 4; ++fm)
#pragma unroll
        for (int fn = 0; fn < 4; ++fn) {
            const int n = bn + wn * 64 + fn * 16 + lrow;
            const float bn_ = bias[n];
#pragma unroll
            for (int r = 0; r < 8; ++r) {
                const int m = bm + wm * 64 + fm * 16 + half * 8 + r;
                const size_t idx = (size_t)m * N + n;
                float vv = acc[fm][fn][r] + bn_;
                if (outF) {
                    if (residual) vv += residual[idx];
                    outF[idx] = vv;
                } else {
                    outB[idx] = (bf16)vv;
                }
            }
        }
}

// ---------------------------------------------------------------------------
// Per-token head attention: scores[h,g] = scale * <q[h,:], k[g,:]> (8x8),
// softmax over g, out[h,d] = sum_g p[h,g] * v[g,d]. One block per token.
// ---------------------------------------------------------------------------
__global__ __launch_bounds__(256)
void attn_per_token(const bf16* __restrict__ Q, const bf16* __restrict__ K,
                    const bf16* __restrict__ V, bf16* __restrict__ O)
{
    __shared__ float sq[C_DIM];
    __shared__ float sk[C_DIM];
    __shared__ float sv[C_DIM];
    __shared__ float sS[H_HEADS * H_HEADS];
    __shared__ float sP[H_HEADS * H_HEADS];

    const int t   = blockIdx.x;
    const int tid = threadIdx.x;
    const size_t base = (size_t)t * C_DIM;

#pragma unroll
    for (int j = 0; j < 8; ++j) {
        const int i = tid * 8 + j;
        sq[i] = (float)Q[base + i];
        sk[i] = (float)K[base + i];
        sv[i] = (float)V[base + i];
    }
    __syncthreads();

    if (tid < 64) {
        const int h = tid >> 3, g = tid & 7;
        const float* qp = &sq[h * D_HEAD];
        const float* kp = &sk[g * D_HEAD];
        float s = 0.f;
        for (int d = 0; d < D_HEAD; ++d) s += qp[d] * kp[d];
        sS[tid] = s * ATT_SCALE;
    }
    __syncthreads();

    if (tid < 8) {
        float mx = -INFINITY;
#pragma unroll
        for (int g = 0; g < 8; ++g) mx = fmaxf(mx, sS[tid * 8 + g]);
        float e[8], sum = 0.f;
#pragma unroll
        for (int g = 0; g < 8; ++g) { e[g] = __expf(sS[tid * 8 + g] - mx); sum += e[g]; }
        const float inv = 1.0f / sum;
#pragma unroll
        for (int g = 0; g < 8; ++g) sP[tid * 8 + g] = e[g] * inv;
    }
    __syncthreads();

    const int o0 = tid * 8;            // 8 contiguous outputs, all in head h
    const int h  = o0 >> 8;
#pragma unroll
    for (int j = 0; j < 8; ++j) {
        const int d = (o0 + j) & (D_HEAD - 1);
        float a = 0.f;
#pragma unroll
        for (int g = 0; g < 8; ++g) a += sP[h * 8 + g] * sv[g * D_HEAD + d];
        O[base + o0 + j] = (bf16)a;
    }
}

// ---------------------------------------------------------------------------
extern "C" void kernel_launch(void* const* d_in, const int* in_sizes, int n_in,
                              void* d_out, int out_size, void* d_ws, size_t ws_size,
                              hipStream_t stream)
{
    const float* x  = (const float*)d_in[0];
    const float* y  = (const float*)d_in[1];
    const float* Wq = (const float*)d_in[2];
    const float* bq = (const float*)d_in[3];
    const float* Wk = (const float*)d_in[4];
    const float* bk = (const float*)d_in[5];
    const float* Wv = (const float*)d_in[6];
    const float* bv = (const float*)d_in[7];
    const float* Wo = (const float*)d_in[8];
    const float* bo = (const float*)d_in[9];
    float* out = (float*)d_out;

    const size_t TC = (size_t)T_TOK * C_DIM;   // 16Mi elems
    const size_t CC = (size_t)C_DIM * C_DIM;   // 4Mi elems

    // Workspace layout (bf16, ~192 MB):
    //   [0, 32M)      xb   (later reused as attention output ab)
    //   [32M, 64M)    yb
    //   [64M, 96M)    Wq/Wk/Wv/Wo bf16 (8 MB each)
    //   [96M, 192M)   q, k, v bf16
    char* ws = (char*)d_ws;
    bf16* xb  = (bf16*)(ws);
    bf16* yb  = (bf16*)(ws + TC * 2);
    bf16* wqb = (bf16*)(ws + TC * 4);
    bf16* wkb = (bf16*)(ws + TC * 4 + CC * 2);
    bf16* wvb = (bf16*)(ws + TC * 4 + CC * 4);
    bf16* wob = (bf16*)(ws + TC * 4 + CC * 6);
    bf16* qb  = (bf16*)(ws + TC * 4 + CC * 8);
    bf16* kb  = (bf16*)(ws + TC * 6 + CC * 8);
    bf16* vb  = (bf16*)(ws + TC * 8 + CC * 8);
    bf16* ab  = xb;   // xb consumed by Q-GEMM before attention writes here

    // 1) precision downconvert
    cvt_f32_to_bf16<<<2048, 256, 0, stream>>>(x,  xb,  (int)TC);
    cvt_f32_to_bf16<<<2048, 256, 0, stream>>>(y,  yb,  (int)TC);
    cvt_f32_to_bf16<<<512,  256, 0, stream>>>(Wq, wqb, (int)CC);
    cvt_f32_to_bf16<<<512,  256, 0, stream>>>(Wk, wkb, (int)CC);
    cvt_f32_to_bf16<<<512,  256, 0, stream>>>(Wv, wvb, (int)CC);
    cvt_f32_to_bf16<<<512,  256, 0, stream>>>(Wo, wob, (int)CC);

    dim3 gblk(256);
    dim3 ggrid(C_DIM / BN, T_TOK / BM);   // (8, 64)

    // 2) q/k/v projections (bf16 out)
    gemm_bf16_wmma<<<ggrid, gblk, 0, stream>>>(xb, wqb, bq, qb, nullptr, nullptr,
                                               T_TOK, C_DIM, C_DIM);
    gemm_bf16_wmma<<<ggrid, gblk, 0, stream>>>(yb, wkb, bk, kb, nullptr, nullptr,
                                               T_TOK, C_DIM, C_DIM);
    gemm_bf16_wmma<<<ggrid, gblk, 0, stream>>>(yb, wvb, bv, vb, nullptr, nullptr,
                                               T_TOK, C_DIM, C_DIM);

    // 3) per-token head attention
    attn_per_token<<<T_TOK, 256, 0, stream>>>(qb, kb, vb, ab);

    // 4) output projection + bias + residual (f32 out)
    gemm_bf16_wmma<<<ggrid, gblk, 0, stream>>>(ab, wob, bo, nullptr, out, x,
                                               T_TOK, C_DIM, C_DIM);
}
